// ParalESNResCor_21380347199625
// MI455X (gfx1250) — matrix operand order repaired and checked by reading, not compiled, not run on hardware
//
#include <hip/hip_runtime.h>

typedef __attribute__((ext_vector_type(16))) _Float16 v16h;
typedef __attribute__((ext_vector_type(8)))  float    v8f;

#define TAUC   0.5f
#define RPAR   3.9f
#define EPSC   0.3f
#define BETAC  0.15f
#define STEPS  15

#define NDIM   64
#define HDIM   256
#define HCC    16
#define ROWS   16
#define THREADS 256
#define WPACK_ELEMS (HDIM * NDIM)   // 16384 halves per packed weight matrix

// ---------------------------------------------------------------------------
// Prep: convert f32 weights to f16 packed in WMMA B-fragment order.
// Packed layout: [tileIdx][lane 0..31][e 0..15] halves, where for f16 B (32x16):
//   N = lane & 15,  K = e + (e >= 8 ? 8 : 0) + (lane >= 16 ? 8 : 0)  within a K-tile of 32.
// tileIdx = nt*KT + kt  (nt = output-col tile, kt = K tile).
// ---------------------------------------------------------------------------
__global__ __launch_bounds__(256) void prep_pack(
    const float* __restrict__ Wre, const float* __restrict__ Wim,
    const float* __restrict__ Wdp,
    _Float16* __restrict__ preRe, _Float16* __restrict__ preIm,
    _Float16* __restrict__ preDp)
{
    int tid  = blockIdx.x * 256 + threadIdx.x;   // 0 .. 16383
    int e    = tid & 15;
    int lane = (tid >> 4) & 31;
    int ktnt = tid >> 9;                         // 0 .. 31
    int row  = lane & 15;
    int ksel = e + ((e >= 8) ? 8 : 0) + ((lane < 16) ? 0 : 8);

    // W_in_{re,im}: (H=256, N=64); B[k][n] = W_in[n*64 + k]; KT = 2 (K=64)
    {
        int nt = ktnt >> 1, kt = ktnt & 1;
        int n  = nt * 16 + row;
        int k  = kt * 32 + ksel;
        preRe[tid] = (_Float16)Wre[n * NDIM + k];
        preIm[tid] = (_Float16)Wim[n * NDIM + k];
    }
    // dp_W: (N=64, H=256); B[k][n] = dp_W[n*256 + k]; KT = 8 (K=256)
    {
        int nt = ktnt >> 3, kt = ktnt & 7;
        int n  = nt * 16 + row;
        int k  = kt * 32 + ksel;
        preDp[tid] = (_Float16)Wdp[n * HDIM + k];
    }
}

// A fragment gather per CDNA5 16-bit A 16x32 layout:
//   M = lane & 15; element e holds K = k0 + e + (e>=8 ? 8 : 0) + (lane>=16 ? 8 : 0)
__device__ __forceinline__ v16h gatherA_f32(const float* __restrict__ rowp,
                                            int k0, int koff)
{
    v16h a;
#pragma unroll
    for (int e = 0; e < 16; ++e) {
        int k = k0 + koff + e + ((e >= 8) ? 8 : 0);
        a[e] = (_Float16)rowp[k];
    }
    return a;
}

__device__ __forceinline__ v16h gatherA_f16(const _Float16* __restrict__ rowp,
                                            int k0, int koff)
{
    v16h a;
#pragma unroll
    for (int e = 0; e < 16; ++e) {
        int k = k0 + koff + e + ((e >= 8) ? 8 : 0);
        a[e] = rowp[k];
    }
    return a;
}

__device__ __forceinline__ v16h loadB(const _Float16* __restrict__ pack,
                                      int tileIdx, int lane)
{
    return *(const v16h*)(pack + ((size_t)(tileIdx * 32 + lane)) * 16);
}

// ---------------------------------------------------------------------------
// Fused kernel: one workgroup = 16 batch rows, 8 wave32s. All intermediates
// stay in LDS; only x in and out go to HBM (~34 MB total -> BW bound).
// ---------------------------------------------------------------------------
__global__ __launch_bounds__(256) void fused_paralesn(
    const float* __restrict__ x,
    const float* __restrict__ b_re, const float* __restrict__ b_im,
    const float* __restrict__ K_mix, const float* __restrict__ dp_b,
    const float* __restrict__ K_local,
    const float* __restrict__ nca_W1, const float* __restrict__ nca_b1,
    const float* __restrict__ nca_W2, const float* __restrict__ nca_b2,
    const _Float16* __restrict__ preRe, const _Float16* __restrict__ preIm,
    const _Float16* __restrict__ preDp,
    float* __restrict__ out)
{
    __shared__ float    x_lds[ROWS][NDIM];     //  4 KB
    __shared__ float    hre  [ROWS][HDIM];     // 16 KB
    __shared__ float    him  [ROWS][HDIM];     // 16 KB
    __shared__ _Float16 mixh [ROWS][HDIM];     //  8 KB
    __shared__ float    drv  [ROWS][NDIM];     //  4 KB
    __shared__ float    gbuf [ROWS][NDIM];     //  4 KB
    __shared__ float    mbuf [ROWS][NDIM];     //  4 KB
    __shared__ float    sW1[HCC * 6];
    __shared__ float    sW2[HCC];
    __shared__ float    sB1[HCC];
    __shared__ float    sKmix[10];
    __shared__ float    sKloc[3];

    const int tid  = threadIdx.x;
    const int lane = tid & 31;
    const int wave = tid >> 5;
    const int row0 = blockIdx.x * ROWS;

    const int mrow  = lane & 15;                // A-matrix row (M)
    const int koff  = (lane >= 16) ? 8 : 0;     // A-matrix lane K offset
    const int mbase = (lane >= 16) ? 8 : 0;     // C/D row group
    const int ncol  = lane & 15;                // C/D column within tile

    // ---- stage 0: load x tile + small constant tensors -------------------
    for (int i = tid; i < ROWS * NDIM; i += THREADS) {
        int r = i >> 6, c = i & 63;
        x_lds[r][c] = x[(size_t)(row0 + r) * NDIM + c];
    }
    if (tid < HCC * 6) sW1[tid] = nca_W1[tid];
    if (tid < HCC)     { sW2[tid] = nca_W2[tid]; sB1[tid] = nca_b1[tid]; }
    if (tid < 10)      sKmix[tid] = K_mix[tid];
    if (tid < 3)       sKloc[tid] = K_local[tid];
    __syncthreads();

    // ---- stage 1: h_re / h_im GEMMs (K=64 as 2 WMMA k-steps) -------------
    {
        v16h a0 = gatherA_f32(&x_lds[mrow][0], 0,  koff);
        v16h a1 = gatherA_f32(&x_lds[mrow][0], 32, koff);
#pragma unroll
        for (int s = 0; s < 2; ++s) {
            int nt = wave * 2 + s;           // 16 column tiles across 8 waves
            int n  = nt * 16 + ncol;
            // real part
            {
                v16h b0 = loadB(preRe, nt * 2 + 0, lane);
                v16h b1 = loadB(preRe, nt * 2 + 1, lane);
                v8f  c  = {};
                c = __builtin_amdgcn_wmma_f32_16x16x32_f16(false, a0, false, b0,
                                                           (short)0, c, false, false);
                c = __builtin_amdgcn_wmma_f32_16x16x32_f16(false, a1, false, b1,
                                                           (short)0, c, false, false);
                float bias = b_re[n];
#pragma unroll
                for (int r2 = 0; r2 < 8; ++r2)
                    hre[mbase + r2][n] = TAUC * (c[r2] + bias);
            }
            // imaginary part
            {
                v16h b0 = loadB(preIm, nt * 2 + 0, lane);
                v16h b1 = loadB(preIm, nt * 2 + 1, lane);
                v8f  c  = {};
                c = __builtin_amdgcn_wmma_f32_16x16x32_f16(false, a0, false, b0,
                                                           (short)0, c, false, false);
                c = __builtin_amdgcn_wmma_f32_16x16x32_f16(false, a1, false, b1,
                                                           (short)0, c, false, false);
                float bias = b_im[n];
#pragma unroll
                for (int r2 = 0; r2 < 8; ++r2)
                    him[mbase + r2][n] = TAUC * (c[r2] + bias);
            }
        }
    }
    __syncthreads();

    // ---- stage 2: circular width-5 mix conv + tanh, stored f16 -----------
    for (int i = tid; i < ROWS * HDIM; i += THREADS) {
        int r = i >> 8, n = i & 255;
        float s = 0.0f;
#pragma unroll
        for (int t = 0; t < 5; ++t) {
            int j = (n + t - 2) & 255;
            s += sKmix[t] * hre[r][j] + sKmix[5 + t] * him[r][j];
        }
        mixh[r][n] = (_Float16)tanhf(s);
    }
    __syncthreads();

    // ---- stage 3: drive GEMM (16x256 @ 256x64), 4 col tiles on waves 0-3 -
    if (wave < 4) {
        int nt = wave;
        v8f c = {};
#pragma unroll
        for (int kt = 0; kt < 8; ++kt) {
            v16h a = gatherA_f16(&mixh[mrow][0], kt * 32, koff);
            v16h b = loadB(preDp, nt * 8 + kt, lane);
            c = __builtin_amdgcn_wmma_f32_16x16x32_f16(false, a, false, b,
                                                       (short)0, c, false, false);
        }
        int   n    = nt * 16 + ncol;
        float bias = dp_b[n];
#pragma unroll
        for (int r2 = 0; r2 < 8; ++r2) {
            float d = (c[r2] + bias + 1.0f) * 0.5f;
            drv[mbase + r2][n]  = d;
            gbuf[mbase + r2][n] = d;
        }
    }
    __syncthreads();

    // ---- stage 4: coupled logistic map, 15 steps, wrap neighbors ---------
    const float kl0 = sKloc[0], kl1 = sKloc[1], kl2 = sKloc[2];
    for (int step = 0; step < STEPS; ++step) {
        for (int i = tid; i < ROWS * NDIM; i += THREADS) {
            int r = i >> 6, cix = i & 63;
            float g = gbuf[r][cix];
            mbuf[r][cix] = RPAR * g * (1.0f - g);
        }
        __syncthreads();
        for (int i = tid; i < ROWS * NDIM; i += THREADS) {
            int r = i >> 6, cix = i & 63;
            float m   = mbuf[r][cix];
            float lft = mbuf[r][(cix + 63) & 63];
            float rgt = mbuf[r][(cix + 1) & 63];
            float local   = kl0 * lft + kl1 * m + kl2 * rgt;
            float physics = (1.0f - EPSC) * m + EPSC * local;
            gbuf[r][cix]  = (1.0f - BETAC) * physics + BETAC * drv[r][cix];
        }
        __syncthreads();
    }

    // ---- stage 5: NCA correction conv + clipped output -------------------
    const float bias2 = nca_b2[0];
    for (int i = tid; i < ROWS * NDIM; i += THREADS) {
        int r = i >> 6, n = i & 63;
        int nm = (n + 63) & 63, np = (n + 1) & 63;
        float d0 = drv[r][nm], d1 = drv[r][n], d2 = drv[r][np];
        float c0 = fminf(fmaxf(gbuf[r][nm], 1e-4f), 1.0f - 1e-4f);
        float c1 = fminf(fmaxf(gbuf[r][n],  1e-4f), 1.0f - 1e-4f);
        float c2 = fminf(fmaxf(gbuf[r][np], 1e-4f), 1.0f - 1e-4f);
        float corr = bias2;
#pragma unroll
        for (int h = 0; h < HCC; ++h) {
            const float* w = &sW1[h * 6];
            float s = sB1[h]
                    + d0 * w[0] + d1 * w[1] + d2 * w[2]
                    + c0 * w[3] + c1 * w[4] + c2 * w[5];
            corr += sW2[h] * fmaxf(s, 0.0f);
        }
        float o = c1 + corr;
        out[(size_t)(row0 + r) * NDIM + n] = fminf(fmaxf(o, 0.0f), 1.0f);
    }
}

extern "C" void kernel_launch(void* const* d_in, const int* in_sizes, int n_in,
                              void* d_out, int out_size, void* d_ws, size_t ws_size,
                              hipStream_t stream)
{
    const float* x     = (const float*)d_in[0];
    const float* Wre   = (const float*)d_in[1];
    const float* Wim   = (const float*)d_in[2];
    const float* bre   = (const float*)d_in[3];
    const float* bim   = (const float*)d_in[4];
    const float* Kmix  = (const float*)d_in[5];
    const float* Wdp   = (const float*)d_in[6];
    const float* dpb   = (const float*)d_in[7];
    const float* Kloc  = (const float*)d_in[8];
    const float* W1    = (const float*)d_in[9];
    const float* b1    = (const float*)d_in[10];
    const float* W2    = (const float*)d_in[11];
    const float* b2    = (const float*)d_in[12];
    float*       outp  = (float*)d_out;

    _Float16* preRe = (_Float16*)d_ws;
    _Float16* preIm = preRe + WPACK_ELEMS;
    _Float16* preDp = preIm + WPACK_ELEMS;

    // Pack weights to f16 fragment layout (runs fresh every call; deterministic).
    prep_pack<<<WPACK_ELEMS / 256, 256, 0, stream>>>(Wre, Wim, Wdp,
                                                     preRe, preIm, preDp);

    const int batch = in_sizes[0] / NDIM;      // 65536
    fused_paralesn<<<batch / ROWS, THREADS, 0, stream>>>(
        x, bre, bim, Kmix, dpb, Kloc, W1, b1, W2, b2,
        preRe, preIm, preDp, outp);
}
